// MambaPolicy_72928544686640
// MI455X (gfx1250) — matrix operand order repaired and checked by reading, not compiled, not run on hardware
//
#include <hip/hip_runtime.h>
#include <hip/hip_bf16.h>
#include <math.h>

// ---- model dims ----
#define BB   4
#define LL   1024
#define INDIM 128
#define DMODEL 256
#define MDIM 512          // 2*DMODEL
#define NL   4
#define DIN  1024         // 2*MDIM
#define NST  16
#define DTR  32           // MDIM/16
#define KC   4
#define ROWS (BB*LL)      // 4096
#define EPSF 1e-5f

#define USE_ASYNC_LDS 1   // global_load_async_to_lds_b32 path in the scan kernel

typedef __attribute__((ext_vector_type(16))) _Float16 v16h;
typedef __attribute__((ext_vector_type(8)))  float    v8f;

// ============================================================================
// Generic WMMA GEMM:  C[M,N] = act( A[M,K] @ W[N,K]^T + bias )
// A fp32 (row stride lda), W fp32 dense (N x K), C fp32 (row stride ldc).
// Requires M%64==0, N%64==0, K%32==0 (true for every call below).
// Block = 128 threads = 4 waves in a 2x2 grid; block tile 64x64; each wave
// owns a 32x32 sub-tile = 4 x v_wmma_f32_16x16x32_f16 per 32-K step.
// Register double-buffering: next K-tile's global loads are issued before the
// WMMAs of the current tile, so s_wait_loadcnt overlaps matrix math.
// act: 0=none, 1=relu, 2=softplus
// ============================================================================
__global__ __launch_bounds__(128)
void wmma_gemm_kernel(const float* __restrict__ A, int lda,
                      const float* __restrict__ W,
                      const float* __restrict__ bias,
                      float* __restrict__ C, int ldc,
                      int M, int N, int K, int act)
{
    __shared__ _Float16 As[64][40];   // 32 K + 8 pad halves
    __shared__ _Float16 Ws[64][40];

    const int t    = threadIdx.x;
    const int wave = t >> 5;
    const int lane = t & 31;
    const int m0   = blockIdx.y * 64;
    const int n0   = blockIdx.x * 64;
    const int wm   = (wave >> 1) * 32;      // wave row offset in tile
    const int wn   = (wave & 1) * 32;       // wave col offset in tile

    // staging ownership: rows r0, r0+16, r0+32, r0+48 ; k-chunk kc..kc+3
    const int r0 = t >> 3;                  // 0..15
    const int kc = (t & 7) << 2;            // 0,4,...,28

    const float* Aptr = A + (size_t)(m0 + r0) * lda + kc;
    const float* Wptr = W + (size_t)(n0 + r0) * K   + kc;
    const size_t astep = (size_t)16 * lda;
    const size_t wstep = (size_t)16 * K;

    float4 areg[4], wreg[4];
#pragma unroll
    for (int i = 0; i < 4; ++i) {           // prologue: K-tile 0
        areg[i] = *reinterpret_cast<const float4*>(Aptr + i * astep);
        wreg[i] = *reinterpret_cast<const float4*>(Wptr + i * wstep);
    }

    v8f acc[2][2] = {};

    for (int k0 = 0; k0 < K; k0 += 32) {
        // convert staged registers -> LDS (f32 -> f16)
#pragma unroll
        for (int i = 0; i < 4; ++i) {
            const int rr = r0 + i * 16;
            As[rr][kc + 0] = (_Float16)areg[i].x;
            As[rr][kc + 1] = (_Float16)areg[i].y;
            As[rr][kc + 2] = (_Float16)areg[i].z;
            As[rr][kc + 3] = (_Float16)areg[i].w;
            Ws[rr][kc + 0] = (_Float16)wreg[i].x;
            Ws[rr][kc + 1] = (_Float16)wreg[i].y;
            Ws[rr][kc + 2] = (_Float16)wreg[i].z;
            Ws[rr][kc + 3] = (_Float16)wreg[i].w;
        }
        __syncthreads();

        // issue next K-tile's loads now; the wait lands after the WMMAs
        if (k0 + 32 < K) {
#pragma unroll
            for (int i = 0; i < 4; ++i) {
                areg[i] = *reinterpret_cast<const float4*>(Aptr + k0 + 32 + i * astep);
                wreg[i] = *reinterpret_cast<const float4*>(Wptr + k0 + 32 + i * wstep);
            }
            if (k0 + 64 < K) {  // L2 prefetch of the tile after next
                __builtin_prefetch(Aptr + k0 + 64, 0, 0);
                __builtin_prefetch(Wptr + k0 + 64, 0, 0);
            }
        }

        // A fragments (16x32, MxK): lane -> row lane%16, K-halves (lane/16)*8, +16
        const int lr   = lane & 15;
        const int kb8  = (lane >> 4) * 8;
        const int kb16 = (lane >> 4) * 16;
        v16h a0, a1, b0, b1;
#pragma unroll
        for (int i = 0; i < 8; ++i) {
            a0[i]     = As[wm + lr][kb8 + i];
            a0[i + 8] = As[wm + lr][kb8 + 16 + i];
            a1[i]     = As[wm + 16 + lr][kb8 + i];
            a1[i + 8] = As[wm + 16 + lr][kb8 + 16 + i];
        }
        // B fragments (32x16, KxN): B[k][n] = W[n][k]
#pragma unroll
        for (int i = 0; i < 8; ++i) {
            b0[i]     = Ws[wn + lr][kb8 + i];
            b0[i + 8] = Ws[wn + lr][kb8 + 16 + i];
            b1[i]     = Ws[wn + 16 + lr][kb8 + i];
            b1[i + 8] = Ws[wn + 16 + lr][kb8 + 16 + i];
        }
        acc[0][0] = __builtin_amdgcn_wmma_f32_16x16x32_f16(false, a0, false, b0, (short)0, acc[0][0], false, false);
        acc[0][1] = __builtin_amdgcn_wmma_f32_16x16x32_f16(false, a0, false, b1, (short)0, acc[0][1], false, false);
        acc[1][0] = __builtin_amdgcn_wmma_f32_16x16x32_f16(false, a1, false, b0, (short)0, acc[1][0], false, false);
        acc[1][1] = __builtin_amdgcn_wmma_f32_16x16x32_f16(false, a1, false, b1, (short)0, acc[1][1], false, false);
        __syncthreads();
    }

    // writeback: VGPR r, lane l -> row r + 8*(l/16), col l%16
    const int colL  = lane & 15;
    const int rhalf = (lane >> 4) * 8;
#pragma unroll
    for (int mi = 0; mi < 2; ++mi) {
#pragma unroll
        for (int ni = 0; ni < 2; ++ni) {
            const int gc = n0 + wn + ni * 16 + colL;
            const float bv = bias ? bias[gc] : 0.0f;
#pragma unroll
            for (int r = 0; r < 8; ++r) {
                float v = acc[mi][ni][r] + bv;
                if (act == 1)      v = fmaxf(v, 0.0f);
                else if (act == 2) v = (v > 20.0f) ? v : log1pf(__expf(v));
                C[(size_t)(m0 + wm + mi * 16 + rhalf + r) * ldc + gc] = v;
            }
        }
    }
}

// ============================================================================
// embed finish: R already holds se in cols [0,256); add reward-embed cols
// [256,512), add sinusoidal PE, LayerNorm over 512. One block (256 thr) / row.
// ============================================================================
__device__ __forceinline__ float pe_val(int l, int c)
{
    const float arg = (float)(c & ~1) * (-9.210340371976184f / 512.0f); // -ln(1e4)/MDIM
    const float ang = (float)l * expf(arg);
    return (c & 1) ? cosf(ang) : sinf(ang);
}

__global__ __launch_bounds__(256)
void embed_pe_ln_kernel(const float* __restrict__ rtg,
                        const float* __restrict__ rw, const float* __restrict__ rb,
                        const float* __restrict__ lnw, const float* __restrict__ lnb,
                        float* __restrict__ R)
{
    const int row = blockIdx.x;
    const int l   = row & (LL - 1);
    const int t   = threadIdx.x;
    const size_t base = (size_t)row * MDIM;

    const float rv = rtg[row];
    float e0 = R[base + t] + pe_val(l, t);
    float e1 = rv * rw[t] + rb[t] + pe_val(l, 256 + t);

    __shared__ float sred[2][8];
    const int lane = t & 31, wv = t >> 5;
    float s = e0 + e1, q = e0 * e0 + e1 * e1;
#pragma unroll
    for (int off = 16; off > 0; off >>= 1) {
        s += __shfl_down(s, off, 32);
        q += __shfl_down(q, off, 32);
    }
    if (lane == 0) { sred[0][wv] = s; sred[1][wv] = q; }
    __syncthreads();
    if (t == 0) {
        float ts = 0.f, tq = 0.f;
        for (int i = 0; i < 8; ++i) { ts += sred[0][i]; tq += sred[1][i]; }
        sred[0][0] = ts; sred[1][0] = tq;
    }
    __syncthreads();
    const float mu  = sred[0][0] * (1.0f / MDIM);
    const float var = sred[1][0] * (1.0f / MDIM) - mu * mu;
    const float inv = rsqrtf(var + EPSF);
    R[base + t]       = (e0 - mu) * inv * lnw[t] + lnb[t];
    R[base + 256 + t] = (e1 - mu) * inv * lnw[256 + t] + lnb[256 + t];
}

// ============================================================================
// residual update + RMSNorm: addmode=0: r=hin; addmode=1: r=residual+hin.
// residual<-r ; out = r * rsqrt(mean(r^2)+eps) * w.  One block (256 thr) / row.
// ============================================================================
__global__ __launch_bounds__(256)
void residual_rms_kernel(const float* __restrict__ hin,
                         float* __restrict__ residual,
                         const float* __restrict__ w,
                         float* __restrict__ out, int addmode)
{
    const int row = blockIdx.x;
    const int t   = threadIdx.x;
    const size_t base = (size_t)row * MDIM;

    float e0 = hin[base + t];
    float e1 = hin[base + 256 + t];
    if (addmode) { e0 += residual[base + t]; e1 += residual[base + 256 + t]; }
    residual[base + t]       = e0;
    residual[base + 256 + t] = e1;

    __shared__ float sred[8];
    const int lane = t & 31, wv = t >> 5;
    float q = e0 * e0 + e1 * e1;
#pragma unroll
    for (int off = 16; off > 0; off >>= 1) q += __shfl_down(q, off, 32);
    if (lane == 0) sred[wv] = q;
    __syncthreads();
    if (t == 0) {
        float tq = 0.f;
        for (int i = 0; i < 8; ++i) tq += sred[i];
        sred[0] = tq;
    }
    __syncthreads();
    const float inv = rsqrtf(sred[0] * (1.0f / MDIM) + EPSF);
    out[base + t]       = e0 * inv * w[t];
    out[base + 256 + t] = e1 * inv * w[256 + t];
}

// ============================================================================
// causal depthwise conv (K=4) + SiLU. xx = XZ cols [0,DIN), row stride 2*DIN.
// ============================================================================
__global__ __launch_bounds__(256)
void conv_silu_kernel(const float* __restrict__ XZ,
                      const float* __restrict__ cw, const float* __restrict__ cb,
                      float* __restrict__ XC)
{
    const int idx = blockIdx.x * 256 + threadIdx.x;   // over B*L*DIN = 4M
    const int d = idx & (DIN - 1);
    const int l = (idx >> 10) & (LL - 1);
    const int b = idx >> 20;
    float acc = cb[d];
#pragma unroll
    for (int k = 0; k < KC; ++k) {
        const int ls = l + k - (KC - 1);
        if (ls >= 0)
            acc += cw[d * KC + k] * XZ[(size_t)(b * LL + ls) * (2 * DIN) + d];
    }
    XC[idx] = acc / (1.0f + __expf(-acc));   // silu
}

// ============================================================================
// selective scan: one lane per (b,d); 16 states in registers; B/C staged to
// LDS each timestep via CDNA5 async global->LDS loads (ASYNCcnt).
// grid = B * (DIN/256) = 16 blocks of 256 threads.
// ============================================================================
__global__ __launch_bounds__(256)
void ssm_scan_kernel(const float* __restrict__ DBCp,  // ROWS x 64 (dt|B|C)
                     const float* __restrict__ DTp,   // ROWS x DIN (softplus'd)
                     const float* __restrict__ XCp,   // ROWS x DIN
                     const float* __restrict__ Alog,  // DIN x NST
                     float* __restrict__ YS)          // ROWS x DIN
{
    const int b = blockIdx.x >> 2;
    const int d = (blockIdx.x & 3) * 256 + threadIdx.x;
    const int t = threadIdx.x;

    float Areg[NST], hc[NST];
#pragma unroll
    for (int n = 0; n < NST; ++n) {
        Areg[n] = -__expf(Alog[d * NST + n]);
        hc[n] = 0.0f;
    }

    __shared__ float sBC[2 * NST];            // B in [0,16), C in [16,32)
#if USE_ASYNC_LDS
    const unsigned ldsOff = (unsigned)(uintptr_t)(&sBC[t]); // flat->LDS = low 32b
#endif

    for (int l = 0; l < LL; ++l) {
        const int row = b * LL + l;
        const float* src = DBCp + (size_t)row * 64 + DTR;
#if USE_ASYNC_LDS
        if (t < 2 * NST) {
            // async DMA of one dword into LDS; tracked by ASYNCcnt
            asm volatile("global_load_async_to_lds_b32 %0, %1, off"
                         :: "v"(ldsOff), "v"(src + t) : "memory");
        }
        asm volatile("s_wait_asynccnt 0" ::: "memory");
#else
        if (t < 2 * NST) sBC[t] = src[t];
#endif
        __syncthreads();
        const float dtv = DTp[(size_t)row * DIN + d];
        const float dx  = dtv * XCp[(size_t)row * DIN + d];
        float yv = 0.0f;
#pragma unroll
        for (int n = 0; n < NST; ++n) {
            hc[n] = __expf(dtv * Areg[n]) * hc[n] + dx * sBC[n];
            yv += hc[n] * sBC[NST + n];
        }
        YS[(size_t)row * DIN + d] = yv;
        __syncthreads();
    }
}

// ============================================================================
// gate: YS <- (YS + Dp*XC) * silu(z);  z = XZ cols [DIN, 2*DIN)
// ============================================================================
__global__ __launch_bounds__(256)
void gate_kernel(float* __restrict__ YS, const float* __restrict__ XC,
                 const float* __restrict__ XZ, const float* __restrict__ Dp)
{
    const int idx = blockIdx.x * 256 + threadIdx.x;   // over 4M
    const int d = idx & (DIN - 1);
    const int row = idx >> 10;
    const float z = XZ[(size_t)row * (2 * DIN) + DIN + d];
    const float y = YS[idx] + Dp[d] * XC[idx];
    YS[idx] = y * (z / (1.0f + __expf(-z)));
}

// ============================================================================
// reward head final dot: out[row] = dot(T1[row,0:256], w2) + b2
// ============================================================================
__global__ __launch_bounds__(256)
void reward_head_kernel(const float* __restrict__ T1, const float* __restrict__ w2,
                        const float* __restrict__ b2, float* __restrict__ outr)
{
    const int row = blockIdx.x;
    const int t   = threadIdx.x;
    float p = T1[(size_t)row * 256 + t] * w2[t];
    __shared__ float sred[8];
    const int lane = t & 31, wv = t >> 5;
#pragma unroll
    for (int off = 16; off > 0; off >>= 1) p += __shfl_down(p, off, 32);
    if (lane == 0) sred[wv] = p;
    __syncthreads();
    if (t == 0) {
        float tp = 0.f;
        for (int i = 0; i < 8; ++i) tp += sred[i];
        outr[row] = tp + b2[0];
    }
}

// ============================================================================
static inline void gemm(const float* A, int lda, const float* W, const float* bias,
                        float* C, int ldc, int M, int N, int K, int act,
                        hipStream_t s)
{
    dim3 grid(N / 64, M / 64), block(128);
    wmma_gemm_kernel<<<grid, block, 0, s>>>(A, lda, W, bias, C, ldc, M, N, K, act);
}

extern "C" void kernel_launch(void* const* d_in, const int* in_sizes, int n_in,
                              void* d_out, int out_size, void* d_ws, size_t ws_size,
                              hipStream_t stream)
{
    const float* x     = (const float*)d_in[0];
    const float* rtg   = (const float*)d_in[1];
    const float* sw    = (const float*)d_in[2];
    const float* sb    = (const float*)d_in[3];
    const float* rw    = (const float*)d_in[4];
    const float* rb    = (const float*)d_in[5];
    const float* lnw   = (const float*)d_in[6];
    const float* lnb   = (const float*)d_in[7];
    const float* ipw   = (const float*)d_in[8];    // (NL, 2048, 512)
    const float* cw    = (const float*)d_in[9];    // (NL, 1024, 4)
    const float* cb    = (const float*)d_in[10];   // (NL, 1024)
    const float* xpw   = (const float*)d_in[11];   // (NL, 64, 1024)
    const float* dpw   = (const float*)d_in[12];   // (NL, 1024, 32)
    const float* dpb   = (const float*)d_in[13];   // (NL, 1024)
    const float* Alog  = (const float*)d_in[14];   // (NL, 1024, 16)
    const float* Dpar  = (const float*)d_in[15];   // (NL, 1024)
    const float* opw   = (const float*)d_in[16];   // (NL, 512, 1024)
    const float* nw    = (const float*)d_in[17];   // (NL, 512)
    const float* nfw   = (const float*)d_in[18];
    const float* sow1  = (const float*)d_in[19];
    const float* sob1  = (const float*)d_in[20];
    const float* sow2  = (const float*)d_in[21];
    const float* sob2  = (const float*)d_in[22];
    const float* row1  = (const float*)d_in[23];
    const float* rob1  = (const float*)d_in[24];
    const float* row2  = (const float*)d_in[25];
    const float* rob2  = (const float*)d_in[26];
    float* out = (float*)d_out;

    // workspace layout (floats)
    float* ws  = (float*)d_ws;
    float* R   = ws;                        // residual     ROWS*512
    float* HN  = R   + (size_t)ROWS * 512;  // normed input ROWS*512
    float* H   = HN  + (size_t)ROWS * 512;  // layer output ROWS*512
    float* XZ  = H   + (size_t)ROWS * 512;  // ROWS*2048
    float* XC  = XZ  + (size_t)ROWS * 2048; // ROWS*1024
    float* DBC = XC  + (size_t)ROWS * 1024; // ROWS*64
    float* DT  = DBC + (size_t)ROWS * 64;   // ROWS*1024
    float* YS  = DT  + (size_t)ROWS * 1024; // ROWS*1024
    float* T1  = YS  + (size_t)ROWS * 1024; // ROWS*256

    // 1) state embedding: se = x @ sw.T + sb  -> R cols [0,256)
    gemm(x, INDIM, sw, sb, R, MDIM, ROWS, DMODEL, INDIM, 0, stream);
    // 2) reward embed + PE + LayerNorm -> R
    embed_pe_ln_kernel<<<ROWS, 256, 0, stream>>>(rtg, rw, rb, lnw, lnb, R);

    // 3) Mamba layers
    for (int i = 0; i < NL; ++i) {
        const float* ipw_i  = ipw  + (size_t)i * 2048 * 512;
        const float* cw_i   = cw   + (size_t)i * DIN * KC;
        const float* cb_i   = cb   + (size_t)i * DIN;
        const float* xpw_i  = xpw  + (size_t)i * 64 * DIN;
        const float* dpw_i  = dpw  + (size_t)i * DIN * DTR;
        const float* dpb_i  = dpb  + (size_t)i * DIN;
        const float* Alog_i = Alog + (size_t)i * DIN * NST;
        const float* Dp_i   = Dpar + (size_t)i * DIN;
        const float* opw_i  = opw  + (size_t)i * MDIM * DIN;
        const float* nw_i   = nw   + (size_t)i * MDIM;

        residual_rms_kernel<<<ROWS, 256, 0, stream>>>(
            (i == 0) ? R : H, R, nw_i, HN, (i == 0) ? 0 : 1);

        // xz = hn @ ipw.T           (4096 x 512) x (2048 x 512)^T
        gemm(HN, MDIM, ipw_i, nullptr, XZ, 2 * DIN, ROWS, 2 * DIN, MDIM, 0, stream);
        // causal conv + silu
        conv_silu_kernel<<<(ROWS * DIN) / 256, 256, 0, stream>>>(XZ, cw_i, cb_i, XC);
        // dbc = xc @ xpw.T          (4096 x 1024) x (64 x 1024)^T
        gemm(XC, DIN, xpw_i, nullptr, DBC, 64, ROWS, 64, DIN, 0, stream);
        // dt = softplus(dbc[:, :32] @ dpw.T + dpb)
        gemm(DBC, 64, dpw_i, dpb_i, DT, DIN, ROWS, DIN, DTR, 2, stream);
        // selective scan
        ssm_scan_kernel<<<BB * (DIN / 256), 256, 0, stream>>>(DBC, DT, XC, Alog_i, YS);
        // gate: YS = (YS + Dp*XC) * silu(z)
        gate_kernel<<<(ROWS * DIN) / 256, 256, 0, stream>>>(YS, XC, XZ, Dp_i);
        // h = y @ opw.T             (4096 x 1024) x (512 x 1024)^T
        gemm(YS, DIN, opw_i, nullptr, H, MDIM, ROWS, MDIM, DIN, 0, stream);
    }

    // 4) final residual + RMSNorm
    residual_rms_kernel<<<ROWS, 256, 0, stream>>>(H, R, nfw, HN, 1);

    // 5) state head: relu(HN @ so_w1.T + b1) @ so_w2.T + b2 -> out[0 : 4096*128)
    gemm(HN, MDIM, sow1, sob1, T1, 256, ROWS, 256, MDIM, 1, stream);
    gemm(T1, 256, sow2, sob2, out, INDIM, ROWS, INDIM, 256, 0, stream);

    // 6) reward head: relu(HN @ ro_w1.T + b1) @ ro_w2.T + b2 -> out[4096*128 : +4096)
    gemm(HN, MDIM, row1, rob1, T1, 256, ROWS, 256, MDIM, 1, stream);
    reward_head_kernel<<<ROWS, 256, 0, stream>>>(T1, row2, rob2,
                                                 out + (size_t)ROWS * INDIM);
}
